// STLSTMGNN_5282809775032
// MI455X (gfx1250) — matrix operand order repaired and checked by reading, not compiled, not run on hardware
//
#include <hip/hip_runtime.h>

typedef __bf16 bf16;
typedef __attribute__((ext_vector_type(8)))  __bf16 v8bf;
typedef __attribute__((ext_vector_type(16))) __bf16 v16bf;
typedef __attribute__((ext_vector_type(8)))  float  v8f;

#define N_SEQ 2944   // B*A = 128*23
#define T_LEN 60
#define HID   128
#define GATES 512    // 4*HID
#define XBF_ELTS (N_SEQ * T_LEN * 16)   // 2,826,240

// Branchless activations: v_exp_f32 + v_add + v_rcp, no EXEC-mask branches.
__device__ __forceinline__ float fast_sig(float x) {
  return __builtin_amdgcn_rcpf(1.0f + __expf(-x));
}
__device__ __forceinline__ float fast_tanh(float x) {
  return 2.0f * __builtin_amdgcn_rcpf(1.0f + __expf(-2.0f * x)) - 1.0f;
}

// ---------------- WMMA fragment loaders (bf16 16x16x32) ----------------
// A (16x32): lanes0-15 row=lane, halfs0-7 = K0..7, halfs8-15 = K16..23;
//            lanes16-31 row=lane-16, halfs0-7 = K8..15, halfs8-15 = K24..31.
__device__ __forceinline__ v16bf load_a_frag(const bf16* base, int ld, int lane, int k0) {
  const int row  = lane & 15;
  const int koff = (lane & 16) ? 8 : 0;
  const bf16* p  = base + row * ld + k0 + koff;
  v8bf lo = *(const v8bf*)p;
  v8bf hi = *(const v8bf*)(p + 16);
  v16bf r;
#pragma unroll
  for (int i = 0; i < 8; ++i) { r[i] = lo[i]; r[i + 8] = hi[i]; }
  return r;
}

// B (32x16), W row-major [out][in] so B[k][n] = W[j0+n][k0+k]:
// lanes0-15 col=lane hold K0..7,K16..23; lanes16-31 col=lane-16 hold K8..15,K24..31.
__device__ __forceinline__ v16bf load_b_frag(const bf16* W, int ld, int j0, int k0, int lane) {
  const int j  = j0 + (lane & 15);
  const int kb = k0 + ((lane & 16) ? 8 : 0);
  const bf16* p = W + (size_t)j * ld + kb;
  v8bf lo = *(const v8bf*)p;
  v8bf hi = *(const v8bf*)(p + 16);
  v16bf r;
#pragma unroll
  for (int i = 0; i < 8; ++i) { r[i] = lo[i]; r[i + 8] = hi[i]; }
  return r;
}

// B fragment where only K=0..15 exist (Wih0, ld==16); lanes0-15 K0..7, lanes16-31 K8..15, halfs8-15 zero
__device__ __forceinline__ v16bf load_b_frag_k16(const bf16* W, int j0, int lane) {
  v16bf r;
  const bf16* p = W + (size_t)(j0 + (lane & 15)) * 16 + ((lane & 16) ? 8 : 0);
  v8bf lo = *(const v8bf*)p;
#pragma unroll
  for (int i = 0; i < 8; ++i) { r[i] = lo[i]; r[i + 8] = (bf16)0.0f; }
  return r;
}

// ---------------- prep: fp32 -> bf16 weights + feature relayout, fused biases ----------------
// xbf layout: [n][t][16] bf16 with n = b*23 + a  (matches x_frames of the reference)
__global__ void prep_kernel(const float* feats,
                            const float* Wih0, const float* Whh0, const float* bih0, const float* bhh0,
                            const float* Wih1, const float* Whh1, const float* bih1, const float* bhh1,
                            bf16* wih0b, bf16* whh0b, bf16* wih1b, bf16* whh1b,
                            float* bias0, float* bias1, bf16* xbf) {
  int i = blockIdx.x * 256 + threadIdx.x;
  if (i < 2 * GATES * 16)  wih0b[i] = (bf16)Wih0[i];
  if (i < 2 * GATES * HID) { whh0b[i] = (bf16)Whh0[i]; whh1b[i] = (bf16)Whh1[i]; }
  if (i < 2 * GATES * 256) wih1b[i] = (bf16)Wih1[i];
  if (i < 2 * GATES)       { bias0[i] = bih0[i] + bhh0[i]; bias1[i] = bih1[i] + bhh1[i]; }
  if (i < XBF_ELTS) {
    int n   = i / (T_LEN * 16);
    int rem = i - n * (T_LEN * 16);
    int t   = rem >> 4;
    int d   = rem & 15;
    int b   = n / 23, a = n - b * 23;
    xbf[i] = (bf16)feats[(((size_t)b * T_LEN + t) * 23 + a) * 16 + d];
  }
}

// ---------------- layer-0 BiLSTM (input dim 16) ----------------
// grid = (184, 2[dir]); block = 256 (8 waves); 16 sequences per block.
__global__ __launch_bounds__(256)
void lstm0_kernel(const bf16* __restrict__ xbf, const bf16* __restrict__ Wih,
                  const bf16* __restrict__ Whh, const float* __restrict__ bias,
                  bf16* __restrict__ out0) {
  const int nbase = blockIdx.x * 16;
  const int dir   = blockIdx.y;
  const int tid   = threadIdx.x;
  const int lane  = tid & 31;
  const int j0b   = (tid >> 5) * 64;   // gate-column base owned by this wave

  __shared__ __align__(16) bf16  sh[16 * HID];
  __shared__ __align__(16) float sc[16 * HID];
  __shared__ __align__(16) float sg[16 * GATES];

  for (int i = tid; i < 16 * HID; i += 256) { sh[i] = (bf16)0.0f; sc[i] = 0.0f; }
  __syncthreads();

  const bf16*  WihD  = Wih  + (size_t)dir * GATES * 16;
  const bf16*  WhhD  = Whh  + (size_t)dir * GATES * HID;
  const float* biasD = bias + dir * GATES;

  // Whh^T fragments resident in VGPRs for all 60 steps (16 frags = 128 VGPRs)
  v16bf whhf[4][4];
#pragma unroll
  for (int nt = 0; nt < 4; ++nt)
#pragma unroll
    for (int kt = 0; kt < 4; ++kt)
      whhf[nt][kt] = load_b_frag(WhhD, HID, j0b + nt * 16, kt * 32, lane);
  v16bf wihf[4];
#pragma unroll
  for (int nt = 0; nt < 4; ++nt) wihf[nt] = load_b_frag_k16(WihD, j0b + nt * 16, lane);

  // per-column biases, loop-invariant
  float bv[4];
#pragma unroll
  for (int nt = 0; nt < 4; ++nt) bv[nt] = biasD[j0b + nt * 16 + (lane & 15)];

  // per-lane x-fragment base (row = lane&15 of this block's 16 sequences)
  const int arow = lane & 15;
  const int koff = (lane & 16) ? 8 : 0;
  const bf16* xrow = xbf + ((size_t)(nbase + arow)) * T_LEN * 16 + koff;

  // prefetch x fragment for step 0 (double-buffered across the time loop)
  v8bf xcur = *(const v8bf*)(xrow + (dir ? (T_LEN - 1) : 0) * 16);

  for (int s = 0; s < T_LEN; ++s) {
    const int tt = dir ? (T_LEN - 1 - s) : s;

    v8f acc[4];
#pragma unroll
    for (int nt = 0; nt < 4; ++nt) {
#pragma unroll
      for (int i = 0; i < 8; ++i) acc[nt][i] = bv[nt];
    }
    { // input GEMM: K=16 (zero-padded A/B to one 32-wide K-step)
      v16bf af;
#pragma unroll
      for (int i = 0; i < 8; ++i) { af[i] = xcur[i]; af[i + 8] = (bf16)0.0f; }
#pragma unroll
      for (int nt = 0; nt < 4; ++nt)
        acc[nt] = __builtin_amdgcn_wmma_f32_16x16x32_bf16(false, af, false, wihf[nt],
                                                          (short)0, acc[nt], false, false);
    }
    { // prefetch next step's x while this step computes
      const int sn = (s + 1 < T_LEN) ? s + 1 : s;
      const int tn = dir ? (T_LEN - 1 - sn) : sn;
      xcur = *(const v8bf*)(xrow + tn * 16);
    }
    { // recurrence: K=128 — issue all LDS A-fragment loads, then the WMMA chain
      v16bf afr[4];
#pragma unroll
      for (int kt = 0; kt < 4; ++kt) afr[kt] = load_a_frag(sh, HID, lane, kt * 32);
#pragma unroll
      for (int kt = 0; kt < 4; ++kt)
#pragma unroll
        for (int nt = 0; nt < 4; ++nt)
          acc[nt] = __builtin_amdgcn_wmma_f32_16x16x32_bf16(false, afr[kt], false, whhf[nt][kt],
                                                            (short)0, acc[nt], false, false);
    }
    { // scatter pre-activation gates to LDS (C layout: lane<16 -> M=r, else M=8+r)
      int col = lane & 15, rb = (lane & 16) ? 8 : 0;
#pragma unroll
      for (int nt = 0; nt < 4; ++nt) {
        int j = j0b + nt * 16 + col;
#pragma unroll
        for (int r = 0; r < 8; ++r) sg[(rb + r) * GATES + j] = acc[nt][r];
      }
    }
    __syncthreads();

    { // i/f/g/o update: 16 rows x 128 units, 8 units per thread (branchless)
      int row = tid >> 4, u0 = (tid & 15) * 8;
#pragma unroll
      for (int u = u0; u < u0 + 8; ++u) {
        float ig = sg[row * GATES + u];
        float fg = sg[row * GATES + HID + u];
        float gg = sg[row * GATES + 2 * HID + u];
        float og = sg[row * GATES + 3 * HID + u];
        float c  = fast_sig(fg) * sc[row * HID + u] + fast_sig(ig) * fast_tanh(gg);
        float h  = fast_sig(og) * fast_tanh(c);
        sc[row * HID + u] = c;
        sh[row * HID + u] = (bf16)h;
        out0[(((size_t)(nbase + row)) * T_LEN + tt) * 256 + dir * HID + u] = (bf16)h;
      }
    }
    __syncthreads();
  }
}

// ---------------- layer-1 BiLSTM (input dim 256), only final h needed ----------------
__global__ __launch_bounds__(256)
void lstm1_kernel(const bf16* __restrict__ out0, const bf16* __restrict__ Wih,
                  const bf16* __restrict__ Whh, const float* __restrict__ bias,
                  float* __restrict__ node) {
  const int nbase = blockIdx.x * 16;
  const int dir   = blockIdx.y;
  const int tid   = threadIdx.x;
  const int lane  = tid & 31;
  const int j0b   = (tid >> 5) * 64;

  __shared__ __align__(16) bf16  sh[16 * HID];
  __shared__ __align__(16) float sc[16 * HID];
  __shared__ __align__(16) float sg[16 * GATES];

  for (int i = tid; i < 16 * HID; i += 256) { sh[i] = (bf16)0.0f; sc[i] = 0.0f; }
  __syncthreads();

  const bf16*  WihD  = Wih  + (size_t)dir * GATES * 256;
  const bf16*  WhhD  = Whh  + (size_t)dir * GATES * HID;
  const float* biasD = bias + dir * GATES;

  v16bf whhf[4][4];
#pragma unroll
  for (int nt = 0; nt < 4; ++nt)
#pragma unroll
    for (int kt = 0; kt < 4; ++kt)
      whhf[nt][kt] = load_b_frag(WhhD, HID, j0b + nt * 16, kt * 32, lane);

  float bv[4];
#pragma unroll
  for (int nt = 0; nt < 4; ++nt) bv[nt] = biasD[j0b + nt * 16 + (lane & 15)];

  const int arow = lane & 15;
  const int koff = (lane & 16) ? 8 : 0;
  const bf16* orow = out0 + (((size_t)(nbase + arow)) * T_LEN) * 256 + koff;

  // double-buffered input A fragments (K=256 => 8 fragments of 16 bf16 per lane)
  v16bf acur[8];
  {
    const bf16* p0 = orow + (size_t)(dir ? (T_LEN - 1) : 0) * 256;
#pragma unroll
    for (int kt = 0; kt < 8; ++kt) {
      v8bf lo = *(const v8bf*)(p0 + kt * 32);
      v8bf hi = *(const v8bf*)(p0 + kt * 32 + 16);
#pragma unroll
      for (int i = 0; i < 8; ++i) { acur[kt][i] = lo[i]; acur[kt][i + 8] = hi[i]; }
    }
  }

  for (int s = 0; s < T_LEN; ++s) {
    v8f acc[4];
#pragma unroll
    for (int nt = 0; nt < 4; ++nt) {
#pragma unroll
      for (int i = 0; i < 8; ++i) acc[nt][i] = bv[nt];
    }
    // input GEMM (K=256): A already in registers; B streamed from L2-hot Wih1
#pragma unroll
    for (int kt = 0; kt < 8; ++kt) {
#pragma unroll
      for (int nt = 0; nt < 4; ++nt) {
        v16bf bfr = load_b_frag(WihD, 256, j0b + nt * 16, kt * 32, lane);
        acc[nt] = __builtin_amdgcn_wmma_f32_16x16x32_bf16(false, acur[kt], false, bfr,
                                                          (short)0, acc[nt], false, false);
      }
    }
    // prefetch next step's input fragments while recurrence runs
    v16bf anext[8];
    {
      const int sn = (s + 1 < T_LEN) ? s + 1 : s;
      const int tn = dir ? (T_LEN - 1 - sn) : sn;
      const bf16* p0 = orow + (size_t)tn * 256;
#pragma unroll
      for (int kt = 0; kt < 8; ++kt) {
        v8bf lo = *(const v8bf*)(p0 + kt * 32);
        v8bf hi = *(const v8bf*)(p0 + kt * 32 + 16);
#pragma unroll
        for (int i = 0; i < 8; ++i) { anext[kt][i] = lo[i]; anext[kt][i + 8] = hi[i]; }
      }
    }
    { // recurrence: K=128 — all LDS A loads first, then WMMA chain
      v16bf afr[4];
#pragma unroll
      for (int kt = 0; kt < 4; ++kt) afr[kt] = load_a_frag(sh, HID, lane, kt * 32);
#pragma unroll
      for (int kt = 0; kt < 4; ++kt)
#pragma unroll
        for (int nt = 0; nt < 4; ++nt)
          acc[nt] = __builtin_amdgcn_wmma_f32_16x16x32_bf16(false, afr[kt], false, whhf[nt][kt],
                                                            (short)0, acc[nt], false, false);
    }
    {
      int col = lane & 15, rb = (lane & 16) ? 8 : 0;
#pragma unroll
      for (int nt = 0; nt < 4; ++nt) {
        int j = j0b + nt * 16 + col;
#pragma unroll
        for (int r = 0; r < 8; ++r) sg[(rb + r) * GATES + j] = acc[nt][r];
      }
    }
    __syncthreads();

    {
      int row = tid >> 4, u0 = (tid & 15) * 8;
#pragma unroll
      for (int u = u0; u < u0 + 8; ++u) {
        float ig = sg[row * GATES + u];
        float fg = sg[row * GATES + HID + u];
        float gg = sg[row * GATES + 2 * HID + u];
        float og = sg[row * GATES + 3 * HID + u];
        float c  = fast_sig(fg) * sc[row * HID + u] + fast_sig(ig) * fast_tanh(gg);
        float h  = fast_sig(og) * fast_tanh(c);
        sc[row * HID + u] = c;
        sh[row * HID + u] = (bf16)h;
        if (s == T_LEN - 1) { // final hidden -> node_repr [B,A,256] (f32)
          int n = nbase + row;
          node[(size_t)n * 256 + dir * HID + u] = h;
        }
      }
    }
    __syncthreads();

#pragma unroll
    for (int kt = 0; kt < 8; ++kt) acur[kt] = anext[kt];
  }
}

// ---------------- GCN head: adj_n is uniform 1/23 => mean over agents ----------------
__global__ __launch_bounds__(128)
void head_kernel(const float* __restrict__ node, const float* __restrict__ W0,
                 const float* __restrict__ b0, const float* __restrict__ W1,
                 const float* __restrict__ b1, const float* __restrict__ cw,
                 const float* __restrict__ cb, float* __restrict__ out) {
  const int b = blockIdx.x, t = threadIdx.x;
  __shared__ float m[256], y0[128], red[128];
  for (int d = t; d < 256; d += 128) {
    float s = 0.0f;
    for (int a = 0; a < 23; ++a) s += node[((size_t)b * 23 + a) * 256 + d];
    m[d] = s * (1.0f / 23.0f);
  }
  __syncthreads();
  float a0 = b0[t];
  for (int d = 0; d < 256; ++d) a0 += m[d] * W0[d * 128 + t];
  y0[t] = fmaxf(a0, 0.0f);
  __syncthreads();
  float a1 = b1[t];
  for (int d = 0; d < 128; ++d) a1 += y0[d] * W1[d * 128 + t];
  red[t] = fmaxf(a1, 0.0f) * cw[t];
  __syncthreads();
  for (int sft = 64; sft > 0; sft >>= 1) {
    if (t < sft) red[t] += red[t + sft];
    __syncthreads();
  }
  if (t == 0) out[b] = red[0] + cb[0];
}

// ---------------- launch ----------------
extern "C" void kernel_launch(void* const* d_in, const int* in_sizes, int n_in,
                              void* d_out, int out_size, void* d_ws, size_t ws_size,
                              hipStream_t stream) {
  const float* feats = (const float*)d_in[0];
  const float* Wih0  = (const float*)d_in[1];
  const float* Whh0  = (const float*)d_in[2];
  const float* bih0  = (const float*)d_in[3];
  const float* bhh0  = (const float*)d_in[4];
  const float* Wih1  = (const float*)d_in[5];
  const float* Whh1  = (const float*)d_in[6];
  const float* bih1  = (const float*)d_in[7];
  const float* bhh1  = (const float*)d_in[8];
  const float* gW0   = (const float*)d_in[9];
  const float* gb0   = (const float*)d_in[10];
  const float* gW1   = (const float*)d_in[11];
  const float* gb1   = (const float*)d_in[12];
  const float* cW    = (const float*)d_in[13];
  const float* cb    = (const float*)d_in[14];

  char* ws = (char*)d_ws;
  bf16*  wih0b = (bf16*)(ws + 0);          //    32768 B
  bf16*  whh0b = (bf16*)(ws + 32768);      //   262144 B
  bf16*  wih1b = (bf16*)(ws + 294912);     //   524288 B
  bf16*  whh1b = (bf16*)(ws + 819200);     //   262144 B
  float* bias0 = (float*)(ws + 1081344);   //     4096 B
  float* bias1 = (float*)(ws + 1085440);   //     4096 B
  float* node  = (float*)(ws + 1089536);   //  3014656 B
  bf16*  out0  = (bf16*)(ws + 4104192);    // 90439680 B
  bf16*  xbf   = (bf16*)(ws + 94543872);   //  5652480 B  (~100.2 MB total)

  const int prep_blocks = (XBF_ELTS + 255) / 256;  // covers the largest range (features)
  prep_kernel<<<prep_blocks, 256, 0, stream>>>(feats, Wih0, Whh0, bih0, bhh0,
                                               Wih1, Whh1, bih1, bhh1,
                                               wih0b, whh0b, wih1b, whh1b,
                                               bias0, bias1, xbf);
  lstm0_kernel<<<dim3(N_SEQ / 16, 2), 256, 0, stream>>>(xbf, wih0b, whh0b, bias0, out0);
  lstm1_kernel<<<dim3(N_SEQ / 16, 2), 256, 0, stream>>>(out0, wih1b, whh1b, bias1, node);
  head_kernel<<<128, 128, 0, stream>>>(node, gW0, gb0, gW1, gb1, cW, cb, (float*)d_out);
}
